// TriModalExpertModel_88665304859115
// MI455X (gfx1250) — compile-verified, matrix operand order checked by reading, not run on hardware
//
#include <hip/hip_runtime.h>

// ---------------------------------------------------------------------------
// Problem constants (fixed by the reference)
// ---------------------------------------------------------------------------
#define NN   4096   // tokens
#define DD   512    // input dim
#define HH   256    // hidden
#define OO   256    // expert out
#define ME   8      // experts
#define FD   512    // fusion dim
#define CC   2      // classes
#define TRIP 768    // 3*OO

typedef __attribute__((ext_vector_type(16))) __bf16 bf16x16;
typedef __attribute__((ext_vector_type(8)))  __bf16 bf16x8;
typedef __attribute__((ext_vector_type(8)))  float  f32x8;

// ---------------------------------------------------------------------------
// fp32 -> bf16 convert (contiguous)
// ---------------------------------------------------------------------------
__global__ __launch_bounds__(256)
void cvt_bf16_kernel(const float* __restrict__ in, __bf16* __restrict__ out, long n) {
    long t = (long)blockIdx.x * 256 + threadIdx.x;
    if (t < n) out[t] = (__bf16)in[t];
}

// ---------------------------------------------------------------------------
// LDS-tiled transpose + convert:  in f32 [z][R][C]  ->  out bf16 [z][C][R]
// grid: (C/32, R/32, Z), block: (32, 8)
// ---------------------------------------------------------------------------
__global__ __launch_bounds__(256)
void transpose_cvt_kernel(const float* __restrict__ in, __bf16* __restrict__ out,
                          int R, int C, long biStride, long boStride) {
    __shared__ float tile[32][33];
    int z  = blockIdx.z;
    int c0 = blockIdx.x * 32;
    int r0 = blockIdx.y * 32;
    for (int i = threadIdx.y; i < 32; i += 8) {
        int r = r0 + i, c = c0 + threadIdx.x;
        tile[i][threadIdx.x] = (r < R && c < C) ? in[z * biStride + (long)r * C + c] : 0.0f;
    }
    __syncthreads();
    for (int i = threadIdx.y; i < 32; i += 8) {
        int c = c0 + i, r = r0 + threadIdx.x;
        if (c < C && r < R)
            out[z * boStride + (long)c * R + r] = (__bf16)tile[threadIdx.x][i];
    }
}

// ---------------------------------------------------------------------------
// Batched bf16 WMMA GEMM:  out[z] = act( A[z/aDiv] (rows x K) * Bt[z%bMod]^T + bias )
//   A  : bf16 row-major [rows x kdim]
//   Bt : bf16 row-major [cols x kdim]   (i.e. B transposed so K is contiguous)
// Each 256-thread block = 8 waves; wave -> 32x32 output tile (2x2 WMMA frags).
// Block tile: 128 rows x 64 cols.  rows/cols/kdim all multiples of tile sizes.
// ---------------------------------------------------------------------------
__global__ __launch_bounds__(256)
void wmma_gemm_bf16(const __bf16* __restrict__ A,  long aStride, int aDiv,
                    const __bf16* __restrict__ Bt, long bStride, int bMod,
                    const float*  __restrict__ bias, int biasStride,
                    float* __restrict__ outF, __bf16* __restrict__ outB,
                    long oStride, int cols, int kdim, int doRelu) {
    const int z = blockIdx.z;
    const __bf16* __restrict__ Ab = A  + (long)(z / aDiv) * aStride;
    const __bf16* __restrict__ Bb = Bt + (long)(z % bMod) * bStride;
    const float*  __restrict__ bb = bias + (long)(z % bMod) * biasStride;

    const int wave = threadIdx.x >> 5;
    const int lane = threadIdx.x & 31;
    const int l16  = lane & 15;
    const int half = lane >> 4;                    // 0/1: K-halves of the 16-bit layout
    const int rowBase = blockIdx.x * 128 + (wave >> 1) * 32;
    const int colBase = blockIdx.y * 64  + (wave &  1) * 32;

    f32x8 acc00 = {}, acc01 = {}, acc10 = {}, acc11 = {};

    // 16-bit A fragment (16x32): lane l16 = row, elements 0..7 -> K = 8*half..,
    // elements 8..15 -> K = 16+8*half..  (ISA 7.12.2). B mirrored via Bt layout.
    const __bf16* aRow0 = Ab + (long)(rowBase + l16) * kdim + 8 * half;
    const __bf16* aRow1 = aRow0 + (long)16 * kdim;
    const __bf16* bCol0 = Bb + (long)(colBase + l16) * kdim + 8 * half;
    const __bf16* bCol1 = bCol0 + (long)16 * kdim;

    union Frag { bf16x16 w; bf16x8 h[2]; };

    for (int k0 = 0; k0 < kdim; k0 += 32) {
        Frag a0, a1, b0, b1;
        a0.h[0] = *(const bf16x8*)(aRow0 + k0);
        a0.h[1] = *(const bf16x8*)(aRow0 + k0 + 16);
        a1.h[0] = *(const bf16x8*)(aRow1 + k0);
        a1.h[1] = *(const bf16x8*)(aRow1 + k0 + 16);
        b0.h[0] = *(const bf16x8*)(bCol0 + k0);
        b0.h[1] = *(const bf16x8*)(bCol0 + k0 + 16);
        b1.h[0] = *(const bf16x8*)(bCol1 + k0);
        b1.h[1] = *(const bf16x8*)(bCol1 + k0 + 16);

        if (k0 + 32 < kdim) {                      // gfx1250 global_prefetch of next K tile
            __builtin_prefetch(aRow0 + k0 + 32, 0, 1);
            __builtin_prefetch(bCol0 + k0 + 32, 0, 1);
        }

        acc00 = __builtin_amdgcn_wmma_f32_16x16x32_bf16(false, a0.w, false, b0.w, (short)0, acc00, false, false);
        acc01 = __builtin_amdgcn_wmma_f32_16x16x32_bf16(false, a0.w, false, b1.w, (short)0, acc01, false, false);
        acc10 = __builtin_amdgcn_wmma_f32_16x16x32_bf16(false, a1.w, false, b0.w, (short)0, acc10, false, false);
        acc11 = __builtin_amdgcn_wmma_f32_16x16x32_bf16(false, a1.w, false, b1.w, (short)0, acc11, false, false);
    }

    // D layout: element v of acc, lane L -> row = 8*(L>>4)+v, col = L&15.
    f32x8 accs[2][2] = {{acc00, acc01}, {acc10, acc11}};
    for (int rt = 0; rt < 2; ++rt) {
        for (int ct = 0; ct < 2; ++ct) {
            int col = colBase + ct * 16 + l16;
            float bv = bias ? bb[col] : 0.0f;
            for (int v = 0; v < 8; ++v) {
                int row = rowBase + rt * 16 + half * 8 + v;
                float val = accs[rt][ct][v] + bv;
                if (doRelu) val = fmaxf(val, 0.0f);
                long idx = (long)z * oStride + (long)row * cols + col;
                if (outB) outB[idx] = (__bf16)val;
                else      outF[idx] = val;
            }
        }
    }
}

// ---------------------------------------------------------------------------
// Gating logits: logits[mod][n][m] = x[mod][n,:] . Wg[:,m] + bg[m]
// one thread per (mod, n, m); 2*NN*ME threads
// ---------------------------------------------------------------------------
__global__ __launch_bounds__(256)
void gate_logits_kernel(const float* __restrict__ xb, const float* __restrict__ xc,
                        const float* __restrict__ Wg, const float* __restrict__ bg,
                        float* __restrict__ logits) {
    int t = blockIdx.x * 256 + threadIdx.x;           // 0 .. 2*NN*ME-1
    if (t >= 2 * NN * ME) return;
    int m   = t & (ME - 1);
    int nm  = t >> 3;
    int n   = nm & (NN - 1);
    int mod = nm >> 12;                               // NN = 4096 = 2^12
    const float* x = (mod == 0 ? xb : xc) + (long)n * DD;
    float s = bg[m];
    for (int d = 0; d < DD; ++d) s += x[d] * Wg[d * ME + m];
    logits[t] = s;
}

// ---------------------------------------------------------------------------
// softmax over 8 experts, top-2 + renorm, dispatch mask; one thread per (mod,n)
// ---------------------------------------------------------------------------
__global__ __launch_bounds__(256)
void softmax_topk_kernel(const float* __restrict__ logits, float* __restrict__ gates,
                         int* __restrict__ idx, float* __restrict__ wts,
                         float* __restrict__ mask) {
    int t = blockIdx.x * 256 + threadIdx.x;           // 0 .. 2*NN-1
    if (t >= 2 * NN) return;
    int mod = t >> 12;
    int n   = t & (NN - 1);
    const float* lg = logits + (long)t * ME;
    float g[ME];
    float mx = lg[0];
    for (int m = 1; m < ME; ++m) mx = fmaxf(mx, lg[m]);
    float s = 0.0f;
    for (int m = 0; m < ME; ++m) { g[m] = __expf(lg[m] - mx); s += g[m]; }
    float inv = 1.0f / s;
    for (int m = 0; m < ME; ++m) { g[m] *= inv; gates[(long)t * ME + m] = g[m]; }
    int i0 = 0;
    for (int m = 1; m < ME; ++m) if (g[m] > g[i0]) i0 = m;
    int i1 = (i0 == 0) ? 1 : 0;
    for (int m = 0; m < ME; ++m) if (m != i0 && g[m] > g[i1]) i1 = m;
    float w0 = g[i0], w1 = g[i1], wsum = w0 + w1;
    idx[(long)t * 2]     = i0;
    idx[(long)t * 2 + 1] = i1;
    wts[(long)t * 2]     = w0 / wsum;
    wts[(long)t * 2 + 1] = w1 / wsum;
    for (int m = 0; m < ME; ++m)
        mask[((long)mod * ME + m) * NN + n] = (m == i0 || m == i1) ? 1.0f : 0.0f;
}

// ---------------------------------------------------------------------------
// Top-2 gather + weighted combine -> bf16 combined [n][768] = [out_b|out_c|out_c]
// y layout: [mod*8+m][NN][OO] f32; one thread per (n, o)
// ---------------------------------------------------------------------------
__global__ __launch_bounds__(256)
void combine_kernel(const float* __restrict__ y, const int* __restrict__ idx,
                    const float* __restrict__ wts, __bf16* __restrict__ combined) {
    long t = (long)blockIdx.x * 256 + threadIdx.x;    // NN*OO
    if (t >= (long)NN * OO) return;
    int o = (int)(t & (OO - 1));
    int n = (int)(t >> 8);

    int  b0 = idx[(long)n * 2], b1 = idx[(long)n * 2 + 1];
    float wb0 = wts[(long)n * 2], wb1 = wts[(long)n * 2 + 1];
    float ob = wb0 * y[((long)b0 * NN + n) * OO + o] +
               wb1 * y[((long)b1 * NN + n) * OO + o];

    long cb = ((long)NN + n) * 2;
    int  c0 = idx[cb], c1 = idx[cb + 1];
    float wc0 = wts[cb], wc1 = wts[cb + 1];
    float oc = wc0 * y[(((long)ME + c0) * NN + n) * OO + o] +
               wc1 * y[(((long)ME + c1) * NN + n) * OO + o];

    __bf16* c = combined + (long)n * TRIP;
    c[o]          = (__bf16)ob;
    c[OO + o]     = (__bf16)oc;
    c[2 * OO + o] = (__bf16)oc;
}

// ---------------------------------------------------------------------------
// classifier: out[n][c] = fused[n,:] . Wc[:,c] + bc[c]; one thread per (n,c)
// ---------------------------------------------------------------------------
__global__ __launch_bounds__(256)
void classifier_kernel(const float* __restrict__ fused, const float* __restrict__ Wc,
                       const float* __restrict__ bc, float* __restrict__ out) {
    int t = blockIdx.x * 256 + threadIdx.x;           // NN*CC
    if (t >= NN * CC) return;
    int c = t & (CC - 1);
    int n = t >> 1;
    const float* f = fused + (long)n * FD;
    float s = bc[c];
    for (int k = 0; k < FD; ++k) s += f[k] * Wc[k * CC + c];
    out[t] = s;
}

// ---------------------------------------------------------------------------
// rho_hat[m] = sum_n g_b + 2*g_c ; rho[m] = sum_n m_b + 2*m_c ; one block per m
// ---------------------------------------------------------------------------
__global__ __launch_bounds__(256)
void rho_kernel(const float* __restrict__ gates, const float* __restrict__ mask,
                float* __restrict__ rho, float* __restrict__ rho_hat) {
    int m = blockIdx.x, tid = threadIdx.x;
    __shared__ float sh1[256], sh2[256];
    float a = 0.0f, b = 0.0f;
    for (int n = tid; n < NN; n += 256) {
        a += gates[(long)n * ME + m] + 2.0f * gates[((long)NN + n) * ME + m];
        b += mask[(long)m * NN + n]  + 2.0f * mask[((long)(ME + m)) * NN + n];
    }
    sh1[tid] = a; sh2[tid] = b;
    __syncthreads();
    for (int s = 128; s > 0; s >>= 1) {
        if (tid < s) { sh1[tid] += sh1[tid + s]; sh2[tid] += sh2[tid + s]; }
        __syncthreads();
    }
    if (tid == 0) { rho_hat[m] = sh1[0]; rho[m] = sh2[0]; }
}

// ---------------------------------------------------------------------------
// sum_out[m][o] = sum_n m_b[m,n]*y_b[m,n,o] + 2*m_c[m,n]*y_c[m,n,o]
// one thread per (m, o): 2048 threads
// ---------------------------------------------------------------------------
__global__ __launch_bounds__(256)
void sum_out_kernel(const float* __restrict__ y, const float* __restrict__ mask,
                    float* __restrict__ sum_out) {
    int t = blockIdx.x * 256 + threadIdx.x;           // ME*OO
    if (t >= ME * OO) return;
    int o = t & (OO - 1);
    int m = t >> 8;
    const float* yb = y + (long)m * NN * OO + o;
    const float* yc = y + (long)(ME + m) * NN * OO + o;
    const float* mb = mask + (long)m * NN;
    const float* mc = mask + (long)(ME + m) * NN;
    float s = 0.0f;
    for (int n = 0; n < NN; ++n)
        s += mb[n] * yb[(long)n * OO] + 2.0f * mc[n] * yc[(long)n * OO];
    sum_out[t] = s;
}

// ---------------------------------------------------------------------------
// finalize: eq_loss and distinctiveness loss; single 256-thread block
// ---------------------------------------------------------------------------
__global__ __launch_bounds__(256)
void finalize_kernel(const float* __restrict__ rho, const float* __restrict__ rho_hat,
                     const float* __restrict__ sum_out, float* __restrict__ out) {
    __shared__ float avg[ME][OO];
    __shared__ float red[256];
    int tid = threadIdx.x;                             // 0..255 == o index
    for (int m = 0; m < ME; ++m)
        avg[m][tid] = sum_out[m * OO + tid] / fmaxf(rho[m], 1.0f);
    __syncthreads();

    float dist_acc = 0.0f;
    int   pair_cnt = 0;
    for (int i = 0; i < ME; ++i) {
        for (int j = i + 1; j < ME; ++j) {
            float d = avg[i][tid] - avg[j][tid];
            red[tid] = d * d;
            __syncthreads();
            for (int s = 128; s > 0; s >>= 1) {
                if (tid < s) red[tid] += red[tid + s];
                __syncthreads();
            }
            bool present = (rho[i] > 0.0f) && (rho[j] > 0.0f);
            if (present) {
                ++pair_cnt;
                dist_acc += __expf(-red[0] * 0.5f);
            }
            __syncthreads();
        }
    }
    if (tid == 0) {
        float eq = 0.0f;
        for (int m = 0; m < ME; ++m) eq += rho[m] * rho_hat[m];
        out[NN * CC]     = (pair_cnt > 0) ? (-dist_acc / (float)pair_cnt) : 0.0f; // dist_loss
        out[NN * CC + 1] = eq / (float)ME;                                        // eq_loss
    }
}

// ---------------------------------------------------------------------------
// launch
// ---------------------------------------------------------------------------
extern "C" void kernel_launch(void* const* d_in, const int* in_sizes, int n_in,
                              void* d_out, int out_size, void* d_ws, size_t ws_size,
                              hipStream_t stream) {
    (void)in_sizes; (void)n_in; (void)out_size; (void)ws_size;

    const float* x_b = (const float*)d_in[0];
    const float* x_c = (const float*)d_in[1];   // fcg branch reuses cfg (source bug)
    const float* Wg  = (const float*)d_in[3];
    const float* bg  = (const float*)d_in[4];
    const float* W1  = (const float*)d_in[5];
    const float* b1  = (const float*)d_in[6];
    const float* W2  = (const float*)d_in[7];
    const float* b2  = (const float*)d_in[8];
    const float* Wf  = (const float*)d_in[9];
    const float* bfv = (const float*)d_in[10];
    const float* Wc  = (const float*)d_in[11];
    const float* bc  = (const float*)d_in[12];
    float* out = (float*)d_out;

    // workspace carve-up (256B aligned)
    char* base = (char*)d_ws;
    size_t off = 0;
    auto carve = [&](size_t bytes) -> char* {
        char* p = base + off;
        off = (off + bytes + 255) & ~(size_t)255;
        return p;
    };
    __bf16* xbf      = (__bf16*)carve((size_t)2 * NN * DD * 2);       // [2][N][D]
    __bf16* W1t      = (__bf16*)carve((size_t)ME * HH * DD * 2);      // [M][H][D]
    __bf16* W2t      = (__bf16*)carve((size_t)ME * OO * HH * 2);      // [M][O][H]
    __bf16* Wft      = (__bf16*)carve((size_t)FD * TRIP * 2);         // [F][3O]
    __bf16* hbuf     = (__bf16*)carve((size_t)2 * ME * NN * HH * 2);  // [16][N][H]
    float*  ybuf     = (float*) carve((size_t)2 * ME * NN * OO * 4);  // [16][N][O]
    float*  logits   = (float*) carve((size_t)2 * NN * ME * 4);
    float*  gates    = (float*) carve((size_t)2 * NN * ME * 4);
    int*    idxb     = (int*)   carve((size_t)2 * NN * 2 * 4);
    float*  wtsb     = (float*) carve((size_t)2 * NN * 2 * 4);
    float*  maskb    = (float*) carve((size_t)2 * ME * NN * 4);       // [2][M][N]
    __bf16* combined = (__bf16*)carve((size_t)NN * TRIP * 2);
    float*  fused    = (float*) carve((size_t)NN * FD * 4);
    float*  rho      = (float*) carve(ME * 4);
    float*  rho_hat  = (float*) carve(ME * 4);
    float*  sum_out  = (float*) carve((size_t)ME * OO * 4);

    const long nND = (long)NN * DD;

    // 1) activations -> bf16
    cvt_bf16_kernel<<<(nND + 255) / 256, 256, 0, stream>>>(x_b, xbf, nND);
    cvt_bf16_kernel<<<(nND + 255) / 256, 256, 0, stream>>>(x_c, xbf + nND, nND);

    // 2) weight transposes (+ bf16 convert) so K is contiguous for WMMA B
    transpose_cvt_kernel<<<dim3(HH / 32, DD / 32, ME), dim3(32, 8), 0, stream>>>(
        W1, W1t, DD, HH, (long)DD * HH, (long)HH * DD);                // [D,H]->[H,D]
    transpose_cvt_kernel<<<dim3(OO / 32, HH / 32, ME), dim3(32, 8), 0, stream>>>(
        W2, W2t, HH, OO, (long)HH * OO, (long)OO * HH);                // [H,O]->[O,H]
    transpose_cvt_kernel<<<dim3(FD / 32, TRIP / 32, 1), dim3(32, 8), 0, stream>>>(
        Wf, Wft, TRIP, FD, 0, 0);                                      // [3O,F]->[F,3O]

    // 3) gating
    gate_logits_kernel<<<(2 * NN * ME) / 256, 256, 0, stream>>>(x_b, x_c, Wg, bg, logits);
    softmax_topk_kernel<<<(2 * NN) / 256, 256, 0, stream>>>(logits, gates, idxb, wtsb, maskb);

    // 4) expert fc1: h = relu(x @ W1[m] + b1[m])  -> bf16, batched over z=mod*8+m
    wmma_gemm_bf16<<<dim3(NN / 128, HH / 64, 2 * ME), 256, 0, stream>>>(
        xbf, nND, ME,                      // A batch = z/8 (modality)
        W1t, (long)HH * DD, ME,            // B batch = z%8 (expert)
        b1, HH,
        nullptr, hbuf, (long)NN * HH, HH, DD, /*relu=*/1);

    // 5) expert fc2: y = h @ W2[m] + b2[m] -> f32
    wmma_gemm_bf16<<<dim3(NN / 128, OO / 64, 2 * ME), 256, 0, stream>>>(
        hbuf, (long)NN * HH, 1,            // A batch = z
        W2t, (long)OO * HH, ME,            // B batch = z%8
        b2, OO,
        ybuf, nullptr, (long)NN * OO, OO, HH, /*relu=*/0);

    // 6) top-2 combine -> bf16 combined [N][768]
    combine_kernel<<<((long)NN * OO) / 256, 256, 0, stream>>>(ybuf, idxb, wtsb, combined);

    // 7) fusion: fused = relu(combined @ Wf + bf) -> f32 [N][512]
    wmma_gemm_bf16<<<dim3(NN / 128, FD / 64, 1), 256, 0, stream>>>(
        combined, 0, 1,
        Wft, 0, 1,
        bfv, 0,
        fused, nullptr, 0, FD, TRIP, /*relu=*/1);

    // 8) classifier -> d_out[0 .. N*C-1]
    classifier_kernel<<<(NN * CC) / 256, 256, 0, stream>>>(fused, Wc, bc, out);

    // 9) losses
    rho_kernel<<<ME, 256, 0, stream>>>(gates, maskb, rho, rho_hat);
    sum_out_kernel<<<(ME * OO) / 256, 256, 0, stream>>>(ybuf, maskb, sum_out);
    finalize_kernel<<<1, 256, 0, stream>>>(rho, rho_hat, sum_out, out);
}
